// DeformableBlock_44083544326619
// MI455X (gfx1250) — matrix operand order repaired and checked by reading, not compiled, not run on hardware
//
#include <hip/hip_runtime.h>

typedef __attribute__((ext_vector_type(16))) _Float16 v16h;
typedef __attribute__((ext_vector_type(8)))  _Float16 v8h;
typedef __attribute__((ext_vector_type(8)))  float    v8f;

#define Bn 2
#define Cc 256
#define Hh 64
#define Ww 64
#define HWn 4096
#define CK 2304   /* 9 * 256 */

/* ---------------- workspace layout (bytes) ---------------- */
#define XH_OFF    (0u)                       /* f16 NHWC input, 4 MB  */
#define XH2_OFF   (4194304u)                 /* f16 NHWC layer-2 input */
#define Y_OFF     (8388608u)                 /* f32 NHWC gemm output, 8 MB */
#define WD1_OFF   (16777216u)                /* f16 [256][2304] */
#define WD2_OFF   (17956864u)
#define WO1_OFF   (19136512u)                /* f16 [32][2304] (18 used) */
#define WO2_OFF   (19283968u)
#define OFFB_OFF  (19431424u)                /* f32 [B*HW][18] offsets */
#define STATS_OFF (20021248u)                /* f32 [16][2] mean,rstd */
#define ZG_OFF    (20021504u)                /* f16 [256] zero guard row */

/* ---------------- layout prep kernels ---------------- */

__global__ void zero_guard_kernel(_Float16* __restrict__ zg) {
  zg[threadIdx.x] = (_Float16)0.f;
}

__global__ void nchw_to_nhwc_f16_kernel(const float* __restrict__ x,
                                        _Float16* __restrict__ xh) {
  const int idx = blockIdx.x * 256 + threadIdx.x;   // NHWC index
  const int c  = idx & 255;
  const int p  = idx >> 8;
  const int b  = p >> 12;
  const int hw = p & 4095;
  xh[idx] = (_Float16)x[(b * Cc + c) * HWn + hw];
}

__global__ void prep_wd_kernel(const float* __restrict__ dw,   // [O][C][3][3]
                               _Float16* __restrict__ wdt) {   // [O][tap*256+c]
  const int idx = blockIdx.x * 256 + threadIdx.x;   // 256*2304
  const int o   = idx / CK;
  const int k   = idx % CK;
  const int tap = k >> 8;
  const int c   = k & 255;
  wdt[idx] = (_Float16)dw[(o * Cc + c) * 9 + tap];
}

__global__ void prep_wo_kernel(const float* __restrict__ w,    // [18][C][3][3]
                               _Float16* __restrict__ wot) {   // [32][tap*256+c]
  const int idx = blockIdx.x * 256 + threadIdx.x;   // 32*2304
  const int j   = idx / CK;
  const int k   = idx % CK;
  const int tap = k >> 8;
  const int c   = k & 255;
  wot[idx] = (j < 18) ? (_Float16)w[(j * Cc + c) * 9 + tap] : (_Float16)0.f;
}

/* ---------------- offset conv: im2col GEMM, N=32 (18 used) ----------------
   128 threads = 4 waves; each wave owns one 16-position M-tile (same image row).
   A rows read straight from global NHWC f16; OOB rows read the zero guard row
   (single pointer select instead of per-element cndmask). */
__global__ void offset_conv_kernel(const _Float16* __restrict__ xh,
                                   const _Float16* __restrict__ wo,
                                   const float* __restrict__ bias,
                                   const _Float16* __restrict__ zg,
                                   float* __restrict__ off) {
  const int lane  = threadIdx.x & 31;
  const int wave  = threadIdx.x >> 5;
  const int tile  = blockIdx.x * 4 + wave;
  const int pbase = tile * 16;
  const int b     = pbase >> 12;
  const int rem   = pbase & 4095;
  const int yrow  = rem >> 6;
  const int xbase = rem & 63;

  const int row  = lane & 15;     // A-matrix M row for this lane
  const int half = lane >> 4;     // K half (A layout) / K sixteenth (B layout)
  const int px   = xbase + row;

  v8f acc0 = {};
  v8f acc1 = {};

  for (int tap = 0; tap < 9; ++tap) {
    const int ky = tap / 3, kx = tap % 3;
    const int yy = yrow + ky - 1;
    const int xx = px + kx - 1;
    const bool valid = (yy >= 0) && (yy < Hh) && (xx >= 0) && (xx < Ww);
    const _Float16* arow =
        valid ? (xh + ((b * Hh + yy) * Ww + xx) * Cc) : zg;

    for (int chunk = 0; chunk < 8; ++chunk) {
      const int c0 = chunk * 32;
      v16h a;
      {
        const v8h lo = *(const v8h*)(arow + c0 + half * 16);
        const v8h hi = *(const v8h*)(arow + c0 + half * 16 + 8);
#pragma unroll
        for (int i = 0; i < 8; ++i) { a[i] = lo[i]; a[8 + i] = hi[i]; }
      }
      const int kidx = tap * 256 + c0 + half * 16;
      {
        const _Float16* bp = wo + (0 + row) * CK + kidx;
        const v8h b0 = *(const v8h*)(bp);
        const v8h b1 = *(const v8h*)(bp + 8);
        v16h bb;
#pragma unroll
        for (int i = 0; i < 8; ++i) { bb[i] = b0[i]; bb[8 + i] = b1[i]; }
        acc0 = __builtin_amdgcn_wmma_f32_16x16x32_f16(false, a, false, bb,
                                                      (short)0, acc0, false, false);
      }
      {
        const _Float16* bp = wo + (16 + row) * CK + kidx;
        const v8h b0 = *(const v8h*)(bp);
        const v8h b1 = *(const v8h*)(bp + 8);
        v16h bb;
#pragma unroll
        for (int i = 0; i < 8; ++i) { bb[i] = b0[i]; bb[8 + i] = b1[i]; }
        acc1 = __builtin_amdgcn_wmma_f32_16x16x32_f16(false, a, false, bb,
                                                      (short)0, acc1, false, false);
      }
    }
  }

  // C/D layout: VGPR r -> M = r + 8*(lane>=16); N = lane%16 (+ tile offset)
  const int j0 = row;
#pragma unroll
  for (int r = 0; r < 8; ++r) {
    const int p = pbase + r + 8 * half;
    off[p * 18 + j0] = acc0[r] + bias[j0];      // j0 in [0,16) always < 18
    const int j1 = 16 + j0;
    if (j1 < 18) off[p * 18 + j1] = acc1[r] + bias[j1];
  }
}

/* ---------------- deformable conv GEMM ----------------
   One workgroup (128 thr, 4 waves) -> 32x256 output tile (M=32: two 16-row
   A sub-tiles share every B fetch -> 8 WMMA per weight-tile load).
   All 9 taps' bilinear addresses/weights precomputed into LDS once; the
   K loop is flattened to 72 steps (kidx = step*32) with a ping-ponged
   A tile so step n+1's gather/build overlaps step n's WMMAs. */
__global__ void deform_gemm_kernel(const _Float16* __restrict__ xh,
                                   const float* __restrict__ off,
                                   const _Float16* __restrict__ wd,
                                   float* __restrict__ y) {
  __shared__ int   s_addr[9][4][32];
  __shared__ float s_w[9][4][32];
  __shared__ __align__(16) _Float16 As[2][32][32];

  const int tid   = threadIdx.x;
  const int lane  = tid & 31;
  const int wave  = tid >> 5;
  const int pbase = blockIdx.x * 32;       // 32 positions, same image row
  const int b     = pbase >> 12;
  const int rem   = pbase & 4095;
  const int yrow  = rem >> 6;
  const int xbase = rem & 63;              // 0 or 32

  /* ---- precompute bilinear coords for all 9 taps x 32 rows ---- */
  for (int i = tid; i < 9 * 32; i += 128) {
    const int tap = i >> 5;
    const int r   = i & 31;
    const int p   = pbase + r;
    const int px  = xbase + r;
    const float dy = off[p * 18 + 2 * tap];
    const float dx = off[p * 18 + 2 * tap + 1];
    const int ky = tap / 3, kx = tap % 3;
    const float py  = (float)(yrow - 1 + ky) + dy;
    const float pxf = (float)(px - 1 + kx) + dx;
    const float y0f = floorf(py);
    const float x0f = floorf(pxf);
    const float ty = py - y0f;
    const float tx = pxf - x0f;
    const int y0 = (int)y0f, x0 = (int)x0f;
    const int y1 = y0 + 1,  x1 = x0 + 1;
    const float wg[4] = {(1.f - ty) * (1.f - tx), (1.f - ty) * tx,
                         ty * (1.f - tx),         ty * tx};
    const int ys[4] = {y0, y0, y1, y1};
    const int xs[4] = {x0, x1, x0, x1};
#pragma unroll
    for (int j = 0; j < 4; ++j) {
      const bool v = (ys[j] >= 0) && (ys[j] < Hh) && (xs[j] >= 0) && (xs[j] < Ww);
      const int yc = v ? ys[j] : 0;
      const int xc = v ? xs[j] : 0;
      s_addr[tap][j][r] = ((b * Hh + yc) * Ww + xc) * Cc;
      s_w[tap][j][r]    = v ? wg[j] : 0.f;
    }
  }
  __syncthreads();

  /* ---- per-thread A-build role: row (0..31), 8 channels ---- */
  const int brow = tid >> 2;
  const int bci  = (tid & 3) * 8;
  int   cad[4];
  float cw[4];

  auto load_coords = [&](int tap) {
#pragma unroll
    for (int j = 0; j < 4; ++j) {
      cad[j] = s_addr[tap][j][brow];
      cw[j]  = s_w[tap][j][brow];
    }
  };
  auto build = [&](int step, int buf) {
    const int c0 = (step & 7) * 32;
    float a[8] = {0.f, 0.f, 0.f, 0.f, 0.f, 0.f, 0.f, 0.f};
#pragma unroll
    for (int j = 0; j < 4; ++j) {
      const v8h v = *(const v8h*)(xh + cad[j] + c0 + bci);
      const float wj = cw[j];
#pragma unroll
      for (int i = 0; i < 8; ++i) a[i] += wj * (float)v[i];
    }
    v8h o;
#pragma unroll
    for (int i = 0; i < 8; ++i) o[i] = (_Float16)a[i];
    *(v8h*)(&As[buf][brow][bci]) = o;
  };

  v8f acc0[4] = {{}, {}, {}, {}};
  v8f acc1[4] = {{}, {}, {}, {}};

  const int row  = lane & 15;
  const int half = lane >> 4;

  load_coords(0);
  build(0, 0);
  __syncthreads();

  for (int step = 0; step < 72; ++step) {
    const int cur = step & 1;

    // A operands for both M sub-tiles from current LDS buffer
    v16h a0, a1;
    {
      const v8h lo0 = *(const v8h*)(&As[cur][row][half * 16]);
      const v8h hi0 = *(const v8h*)(&As[cur][row][half * 16 + 8]);
      const v8h lo1 = *(const v8h*)(&As[cur][16 + row][half * 16]);
      const v8h hi1 = *(const v8h*)(&As[cur][16 + row][half * 16 + 8]);
#pragma unroll
      for (int i = 0; i < 8; ++i) {
        a0[i] = lo0[i]; a0[8 + i] = hi0[i];
        a1[i] = lo1[i]; a1[8 + i] = hi1[i];
      }
    }

    // overlap: build next step's A tile into the other buffer
    const int nstep = step + 1;
    if (nstep < 72) {
      if ((nstep & 7) == 0) load_coords(nstep >> 3);
      build(nstep, 1 - cur);
    }

    const int kidx = step * 32 + half * 16;
    // prefetch next step's weight row for this lane's first sub-tile column
    __builtin_prefetch(wd + (wave * 64 + row) * CK + kidx + 32, 0, 1);

#pragma unroll
    for (int s = 0; s < 4; ++s) {
      const int oc = wave * 64 + s * 16 + row;
      const _Float16* bp = wd + oc * CK + kidx;
      const v8h b0 = *(const v8h*)(bp);
      const v8h b1 = *(const v8h*)(bp + 8);
      v16h bb;
#pragma unroll
      for (int i = 0; i < 8; ++i) { bb[i] = b0[i]; bb[8 + i] = b1[i]; }
      acc0[s] = __builtin_amdgcn_wmma_f32_16x16x32_f16(false, a0, false, bb,
                                                       (short)0, acc0[s], false, false);
      acc1[s] = __builtin_amdgcn_wmma_f32_16x16x32_f16(false, a1, false, bb,
                                                       (short)0, acc1[s], false, false);
    }
    __syncthreads();
  }

#pragma unroll
  for (int s = 0; s < 4; ++s) {
    const int oc = wave * 64 + s * 16 + row;
#pragma unroll
    for (int r = 0; r < 8; ++r) {
      const int p0 = pbase + r + 8 * half;
      y[p0 * Cc + oc]        = acc0[s][r];
      y[(p0 + 16) * Cc + oc] = acc1[s][r];
    }
  }
}

/* ---------------- GroupNorm ---------------- */

__global__ void gn_stats_kernel(const float* __restrict__ y,
                                float* __restrict__ stats) {
  __shared__ float ssum[256];
  __shared__ float ssq[256];
  const int bg = blockIdx.x;        // 16 = B * groups
  const int b = bg >> 3, g = bg & 7;
  const float* base = y + (size_t)b * HWn * Cc + g * 32;
  float s = 0.f, q = 0.f;
  for (int i = threadIdx.x; i < HWn * 32; i += 256) {
    const int p = i >> 5;
    const int c = i & 31;
    const float v = base[p * Cc + c];
    s += v; q += v * v;
  }
  ssum[threadIdx.x] = s;
  ssq[threadIdx.x]  = q;
  __syncthreads();
  for (int st = 128; st > 0; st >>= 1) {
    if (threadIdx.x < st) {
      ssum[threadIdx.x] += ssum[threadIdx.x + st];
      ssq[threadIdx.x]  += ssq[threadIdx.x + st];
    }
    __syncthreads();
  }
  if (threadIdx.x == 0) {
    const float n  = (float)(HWn * 32);
    const float mu = ssum[0] / n;
    const float var = ssq[0] / n - mu * mu;
    stats[bg * 2]     = mu;
    stats[bg * 2 + 1] = rsqrtf(var + 1e-5f);
  }
}

template <int MODE>   // 0: f16 NHWC (feeds layer 2), 1: f32 NCHW (final output)
__global__ void gn_apply_kernel(const float* __restrict__ y,
                                const float* __restrict__ stats,
                                const float* __restrict__ sc,
                                const float* __restrict__ bi,
                                _Float16* __restrict__ out_h,
                                float* __restrict__ out_f) {
  const int idx = blockIdx.x * 256 + threadIdx.x;  // NHWC index
  const int c  = idx & 255;
  const int p  = idx >> 8;
  const int b  = p >> 12;
  const int g  = c >> 5;
  const int bg = b * 8 + g;
  const float mu = stats[bg * 2];
  const float rs = stats[bg * 2 + 1];
  float v = (y[idx] - mu) * rs * sc[c] + bi[c];
  v = v > 0.f ? v : 0.f;
  if (MODE == 0) {
    out_h[idx] = (_Float16)v;
  } else {
    const int hw = p & 4095;
    out_f[(b * Cc + c) * HWn + hw] = v;
  }
}

/* ---------------- launch ---------------- */

extern "C" void kernel_launch(void* const* d_in, const int* in_sizes, int n_in,
                              void* d_out, int out_size, void* d_ws, size_t ws_size,
                              hipStream_t stream) {
  const float* x      = (const float*)d_in[0];
  const float* off1_w = (const float*)d_in[1];
  const float* off1_b = (const float*)d_in[2];
  const float* dw1    = (const float*)d_in[3];
  const float* gn1_s  = (const float*)d_in[4];
  const float* gn1_b  = (const float*)d_in[5];
  const float* off2_w = (const float*)d_in[6];
  const float* off2_b = (const float*)d_in[7];
  const float* dw2    = (const float*)d_in[8];
  const float* gn2_s  = (const float*)d_in[9];
  const float* gn2_b  = (const float*)d_in[10];
  float* out = (float*)d_out;
  char* ws = (char*)d_ws;

  _Float16* xh    = (_Float16*)(ws + XH_OFF);
  _Float16* xh2   = (_Float16*)(ws + XH2_OFF);
  float*    yb    = (float*)(ws + Y_OFF);
  _Float16* wd1t  = (_Float16*)(ws + WD1_OFF);
  _Float16* wd2t  = (_Float16*)(ws + WD2_OFF);
  _Float16* wo1t  = (_Float16*)(ws + WO1_OFF);
  _Float16* wo2t  = (_Float16*)(ws + WO2_OFF);
  float*    offb  = (float*)(ws + OFFB_OFF);
  float*    stats = (float*)(ws + STATS_OFF);
  _Float16* zg    = (_Float16*)(ws + ZG_OFF);

  const int nElem = Bn * HWn * Cc;  // 2M

  zero_guard_kernel<<<1, 256, 0, stream>>>(zg);
  nchw_to_nhwc_f16_kernel<<<nElem / 256, 256, 0, stream>>>(x, xh);
  prep_wd_kernel<<<(Cc * CK) / 256, 256, 0, stream>>>(dw1, wd1t);
  prep_wd_kernel<<<(Cc * CK) / 256, 256, 0, stream>>>(dw2, wd2t);
  prep_wo_kernel<<<(32 * CK) / 256, 256, 0, stream>>>(off1_w, wo1t);
  prep_wo_kernel<<<(32 * CK) / 256, 256, 0, stream>>>(off2_w, wo2t);

  // layer 1
  offset_conv_kernel<<<128, 128, 0, stream>>>(xh, wo1t, off1_b, zg, offb);
  deform_gemm_kernel<<<256, 128, 0, stream>>>(xh, offb, wd1t, yb);
  gn_stats_kernel<<<16, 256, 0, stream>>>(yb, stats);
  gn_apply_kernel<0><<<nElem / 256, 256, 0, stream>>>(yb, stats, gn1_s, gn1_b, xh2, nullptr);

  // layer 2
  offset_conv_kernel<<<128, 128, 0, stream>>>(xh2, wo2t, off2_b, zg, offb);
  deform_gemm_kernel<<<256, 128, 0, stream>>>(xh2, offb, wd2t, yb);
  gn_stats_kernel<<<16, 256, 0, stream>>>(yb, stats);
  gn_apply_kernel<1><<<nElem / 256, 256, 0, stream>>>(yb, stats, gn2_s, gn2_b, nullptr, out);
}
